// MultiHeadAtt_82755429860250
// MI455X (gfx1250) — matrix operand order repaired and checked by reading, not compiled, not run on hardware
//
#include <hip/hip_runtime.h>
#include <hip/hip_bf16.h>

typedef __bf16 bf16_t;
typedef __attribute__((ext_vector_type(16))) __bf16 v16bf;
typedef __attribute__((ext_vector_type(8)))  __bf16 v8bf;
typedef __attribute__((ext_vector_type(8)))  float  v8f;

static constexpr int Bn = 4, Sn = 2048, Dn = 1024, Hn = 16, DKn = 64;

#define WMMA_BF16(a, b, c) \
  __builtin_amdgcn_wmma_f32_16x16x32_bf16(false, (a), false, (b), (short)0, (c), false, false)

__device__ __forceinline__ v16bf cat8(v8bf lo, v8bf hi) {
  return __builtin_shufflevector(lo, hi, 0,1,2,3,4,5,6,7,8,9,10,11,12,13,14,15);
}

// ---- A fragment (16x32, bf16): lane holds row M = lane&15; K = g*16 + half*8 + i
__device__ __forceinline__ v16bf loadA_bf(const bf16_t* __restrict__ src, int ld,
                                          int m0, int k0, int lane) {
  int half = (lane >> 4) & 1, row = m0 + (lane & 15);
  const bf16_t* p = src + (size_t)row * ld + k0 + half * 8;
  v8bf lo = *(const v8bf*)p;
  v8bf hi = *(const v8bf*)(p + 16);
  return cat8(lo, hi);
}
// ---- B fragment (32x16): lane holds col N = lane&15; K = g*16 + half*8 + i
__device__ __forceinline__ v16bf loadB_bf(const bf16_t* __restrict__ src, int ld,
                                          int n0, int k0, int lane) {
  int half = (lane >> 4) & 1, col = n0 + (lane & 15);
  const bf16_t* p = src + (size_t)col * ld + k0 + half * 8;
  v8bf lo = *(const v8bf*)p;
  v8bf hi = *(const v8bf*)(p + 16);
  return cat8(lo, hi);
}

// ---- flat f32 -> bf16 conversion (4 elems/thread)
__global__ __launch_bounds__(256) void cvt_bf16(const float* __restrict__ src,
                                                bf16_t* __restrict__ dst, int n) {
  int i = (blockIdx.x * 256 + threadIdx.x) * 4;
  if (i + 3 < n) {
#pragma unroll
    for (int j = 0; j < 4; ++j) dst[i + j] = (bf16_t)src[i + j];
  }
}

// ---- LDS-tiled transpose + convert: dst[c][r] = (bf16)src[r][c]; 32x32 tile, 256 threads
__global__ __launch_bounds__(256) void transpose_cvt(const float* __restrict__ src,
                                                     bf16_t* __restrict__ dst,
                                                     int rows, int cols) {
  __shared__ float tile[32][33];
  int c0 = blockIdx.x * 32, r0 = blockIdx.y * 32;
  int tx = threadIdx.x & 31, ty = threadIdx.x >> 5;  // 8 rows per pass
#pragma unroll
  for (int i = 0; i < 32; i += 8)
    tile[ty + i][tx] = src[(size_t)(r0 + ty + i) * cols + c0 + tx];
  __syncthreads();
#pragma unroll
  for (int i = 0; i < 32; i += 8)
    dst[(size_t)(c0 + ty + i) * rows + r0 + tx] = (bf16_t)tile[tx][ty + i];
}

enum { OUT_F32 = 0, OUT_QK = 1, OUT_VT = 2, OUT_BF16 = 3 };

// ---- main bf16 GEMM: Y[m][n] = sum_k A[m][k]*B[n][k]; one wave = 32x64 tile (8 WMMA / k-step)
template <int OMODE>
__global__ __launch_bounds__(256) void gemm_bf16(const bf16_t* __restrict__ A,
                                                 const bf16_t* __restrict__ Bm,
                                                 void* __restrict__ Ov,
                                                 int M, int N, int K) {
  int lane = threadIdx.x & 31, wave = threadIdx.x >> 5;
  int m0 = (blockIdx.x * 8 + wave) * 32;
  int n0 = blockIdx.y * 64;
  v8f acc[2][4];
#pragma unroll
  for (int u = 0; u < 2; ++u)
#pragma unroll
    for (int f = 0; f < 4; ++f)
#pragma unroll
      for (int r = 0; r < 8; ++r) acc[u][f][r] = 0.0f;

  for (int k0 = 0; k0 < K; k0 += 32) {
    v16bf a0 = loadA_bf(A, K, m0, k0, lane);
    v16bf a1 = loadA_bf(A, K, m0 + 16, k0, lane);
#pragma unroll
    for (int f = 0; f < 4; ++f) {
      v16bf b = loadB_bf(Bm, K, n0 + f * 16, k0, lane);
      acc[0][f] = WMMA_BF16(a0, b, acc[0][f]);
      acc[1][f] = WMMA_BF16(a1, b, acc[1][f]);
    }
  }

  int half = (lane >> 4) & 1, nin = lane & 15;
#pragma unroll
  for (int u = 0; u < 2; ++u)
#pragma unroll
    for (int f = 0; f < 4; ++f)
#pragma unroll
      for (int r = 0; r < 8; ++r) {
        int m = m0 + u * 16 + r + half * 8, n = n0 + f * 16 + nin;
        float v = acc[u][f][r];
        if (OMODE == OUT_F32) {
          ((float*)Ov)[(size_t)m * N + n] = v;
        } else if (OMODE == OUT_BF16) {
          ((bf16_t*)Ov)[(size_t)m * N + n] = (bf16_t)v;
        } else {
          int b = m >> 11, s = m & (Sn - 1), h = n >> 6, dk = n & (DKn - 1);
          if (OMODE == OUT_QK)
            ((bf16_t*)Ov)[(((size_t)(b * Hn + h)) * Sn + s) * DKn + dk] = (bf16_t)v;
          else
            ((bf16_t*)Ov)[(((size_t)(b * Hn + h)) * DKn + dk) * Sn + s] = (bf16_t)v;
        }
      }
}

// ---- flash attention: one wave = 16 query rows of one (b,h); key tiles of 32.
__global__ __launch_bounds__(256) void attn_kernel(const bf16_t* __restrict__ q,
                                                   const bf16_t* __restrict__ k,
                                                   const bf16_t* __restrict__ vt,
                                                   const int* __restrict__ mask,
                                                   bf16_t* __restrict__ Hb) {
  __shared__ bf16_t plds[8 * 16 * 40];  // per-wave 16x32 P tile, row stride 40 (16B aligned)
  int lane = threadIdx.x & 31, wave = threadIdx.x >> 5;
  int half = (lane >> 4) & 1, nin = lane & 15;
  int b = blockIdx.z, h = blockIdx.y;
  int q0 = blockIdx.x * 128 + wave * 16;
  const bf16_t* qh = q + ((size_t)(b * Hn + h)) * Sn * DKn;
  const bf16_t* kh = k + ((size_t)(b * Hn + h)) * Sn * DKn;
  const bf16_t* vh = vt + ((size_t)(b * Hn + h)) * DKn * Sn;
  const int* mb = mask + (size_t)b * Sn * Sn;
  bf16_t* myl = plds + wave * 16 * 40;

  v16bf qa0 = loadA_bf(qh, DKn, q0, 0, lane);
  v16bf qa1 = loadA_bf(qh, DKn, q0, 32, lane);

  float mrow[8], lrow[8];
  v8f o[4];
#pragma unroll
  for (int r = 0; r < 8; ++r) { mrow[r] = -__builtin_inff(); lrow[r] = 0.0f; }
#pragma unroll
  for (int f = 0; f < 4; ++f)
#pragma unroll
    for (int r = 0; r < 8; ++r) o[f][r] = 0.0f;

  const float scale = 0.125f;  // 1/sqrt(64)
  for (int n0 = 0; n0 < Sn; n0 += 32) {
    // scores: two 16x16 fragments, each K=64 (2 WMMA steps)
    v8f sc[2];
#pragma unroll
    for (int t = 0; t < 2; ++t) {
      v16bf kb0 = loadB_bf(kh, DKn, n0 + t * 16, 0, lane);
      v16bf kb1 = loadB_bf(kh, DKn, n0 + t * 16, 32, lane);
      v8f c;
#pragma unroll
      for (int r = 0; r < 8; ++r) c[r] = 0.0f;
      c = WMMA_BF16(qa0, kb0, c);
      c = WMMA_BF16(qa1, kb1, c);
      sc[t] = c;
    }
    // scale + mask (reference: masked positions get +inf, verbatim)
    float s[2][8];
#pragma unroll
    for (int t = 0; t < 2; ++t)
#pragma unroll
      for (int r = 0; r < 8; ++r) {
        float v = sc[t][r] * scale;
        int qrow = q0 + r + half * 8, key = n0 + t * 16 + nin;
        if (mb[(size_t)qrow * Sn + key] == 0) v = __builtin_inff();
        s[t][r] = v;
      }
    // online softmax; row reductions across the 16-lane group via shfl_xor
    float p[2][8], alpha[8];
#pragma unroll
    for (int r = 0; r < 8; ++r) {
      float rm = fmaxf(s[0][r], s[1][r]);
#pragma unroll
      for (int off = 1; off < 16; off <<= 1) rm = fmaxf(rm, __shfl_xor(rm, off, 32));
      float mn = fmaxf(mrow[r], rm);
      float al = __expf(mrow[r] - mn);
      float p0 = __expf(s[0][r] - mn), p1 = __expf(s[1][r] - mn);
      p[0][r] = p0; p[1][r] = p1;
      float rs = p0 + p1;
#pragma unroll
      for (int off = 1; off < 16; off <<= 1) rs += __shfl_xor(rs, off, 32);
      lrow[r] = lrow[r] * al + rs;
      mrow[r] = mn;
      alpha[r] = al;
    }
#pragma unroll
    for (int f = 0; f < 4; ++f)
#pragma unroll
      for (int r = 0; r < 8; ++r) o[f][r] *= alpha[r];

    // stage P (C layout -> A layout) through per-wave LDS tile
#pragma unroll
    for (int t = 0; t < 2; ++t)
#pragma unroll
      for (int r = 0; r < 8; ++r)
        myl[(r + half * 8) * 40 + t * 16 + nin] = (bf16_t)p[t][r];
    asm volatile("s_wait_dscnt 0x0" ::: "memory");
    v16bf pa;
    {
      const bf16_t* pp = myl + (lane & 15) * 40 + half * 8;
#pragma unroll
      for (int g = 0; g < 2; ++g)
#pragma unroll
        for (int i = 0; i < 8; ++i) pa[g * 8 + i] = pp[g * 16 + i];
    }
    // O += P @ V  (V stored transposed -> contiguous B loads over keys)
#pragma unroll
    for (int f = 0; f < 4; ++f) {
      v16bf vb = loadB_bf(vh, Sn, f * 16, n0, lane);
      o[f] = WMMA_BF16(pa, vb, o[f]);
    }
  }
  // finalize and scatter heads into concatenated [B,S,D] bf16
#pragma unroll
  for (int f = 0; f < 4; ++f)
#pragma unroll
    for (int r = 0; r < 8; ++r) {
      float v = o[f][r] / lrow[r];
      int srow = q0 + r + half * 8;
      Hb[((size_t)(b * Sn + srow)) * Dn + h * DKn + f * 16 + nin] = (bf16_t)v;
    }
}

extern "C" void kernel_launch(void* const* d_in, const int* in_sizes, int n_in,
                              void* d_out, int out_size, void* d_ws, size_t ws_size,
                              hipStream_t stream) {
  const float* E  = (const float*)d_in[0];
  const float* WQ = (const float*)d_in[1];
  const float* WK = (const float*)d_in[2];
  const float* WV = (const float*)d_in[3];
  const float* WO = (const float*)d_in[4];
  const float* HQ = (const float*)d_in[5];
  const float* HK = (const float*)d_in[6];
  const float* HV = (const float*)d_in[7];
  const int*  msk = (const int*)d_in[8];
  float* out = (float*)d_out;

  char* ws = (char*)d_ws;
  const size_t nE  = (size_t)Bn * Sn * Dn;  // 8.4M
  const size_t nW  = (size_t)Dn * Dn;       // 1M
  const size_t szW = nW * sizeof(bf16_t);   // 2 MB
  const size_t szQKV = (size_t)Bn * Hn * Sn * DKn * sizeof(bf16_t);  // 16 MB
  bf16_t* Ebf  = (bf16_t*)ws; ws += nE * sizeof(bf16_t);  // 16 MB
  bf16_t* WObf = (bf16_t*)ws; ws += szW;
  bf16_t* HbfQ = (bf16_t*)ws; ws += szW;
  bf16_t* HbfK = (bf16_t*)ws; ws += szW;
  bf16_t* HbfV = (bf16_t*)ws; ws += szW;
  bf16_t* WTQ  = (bf16_t*)ws; ws += szW;
  bf16_t* WTK  = (bf16_t*)ws; ws += szW;
  bf16_t* WTV  = (bf16_t*)ws; ws += szW;
  bf16_t* WcQ  = (bf16_t*)ws; ws += szW;
  bf16_t* WcK  = (bf16_t*)ws; ws += szW;
  bf16_t* WcV  = (bf16_t*)ws; ws += szW;
  bf16_t* qb   = (bf16_t*)ws; ws += szQKV;
  bf16_t* kb   = (bf16_t*)ws; ws += szQKV;
  bf16_t* vtb  = (bf16_t*)ws; ws += szQKV;
  bf16_t* Hb   = (bf16_t*)ws; ws += nE * sizeof(bf16_t);

  dim3 blk(256);
  // 0) one-time conversions / transposes to bf16
  hipLaunchKernelGGL(cvt_bf16, dim3((unsigned)(nE / 4 / 256)), blk, 0, stream, E, Ebf, (int)nE);
  hipLaunchKernelGGL(cvt_bf16, dim3((unsigned)(nW / 4 / 256)), blk, 0, stream, WO, WObf, (int)nW);
  hipLaunchKernelGGL(cvt_bf16, dim3((unsigned)(nW / 4 / 256)), blk, 0, stream, HQ, HbfQ, (int)nW);
  hipLaunchKernelGGL(cvt_bf16, dim3((unsigned)(nW / 4 / 256)), blk, 0, stream, HK, HbfK, (int)nW);
  hipLaunchKernelGGL(cvt_bf16, dim3((unsigned)(nW / 4 / 256)), blk, 0, stream, HV, HbfV, (int)nW);
  dim3 gT(Dn / 32, Dn / 32);
  hipLaunchKernelGGL(transpose_cvt, gT, blk, 0, stream, WQ, WTQ, Dn, Dn);
  hipLaunchKernelGGL(transpose_cvt, gT, blk, 0, stream, WK, WTK, Dn, Dn);
  hipLaunchKernelGGL(transpose_cvt, gT, blk, 0, stream, WV, WTV, Dn, Dn);
  // 1) fold per-head weights into the base projection: Wc[n'][d] = sum_e H[n'][e] * W[e][d]
  dim3 gC(Dn / 256, Dn / 64);
  hipLaunchKernelGGL((gemm_bf16<OUT_BF16>), gC, blk, 0, stream, HbfQ, WTQ, (void*)WcQ, Dn, Dn, Dn);
  hipLaunchKernelGGL((gemm_bf16<OUT_BF16>), gC, blk, 0, stream, HbfK, WTK, (void*)WcK, Dn, Dn, Dn);
  hipLaunchKernelGGL((gemm_bf16<OUT_BF16>), gC, blk, 0, stream, HbfV, WTV, (void*)WcV, Dn, Dn, Dn);
  // 2) fused projections (pure bf16, 32x64/wave): q/k -> [B,H,S,DK], v -> [B,H,DK,S]
  dim3 gP((Bn * Sn) / 256, Dn / 64);
  hipLaunchKernelGGL((gemm_bf16<OUT_QK>), gP, blk, 0, stream, Ebf, WcQ, (void*)qb, Bn * Sn, Dn, Dn);
  hipLaunchKernelGGL((gemm_bf16<OUT_QK>), gP, blk, 0, stream, Ebf, WcK, (void*)kb, Bn * Sn, Dn, Dn);
  hipLaunchKernelGGL((gemm_bf16<OUT_VT>), gP, blk, 0, stream, Ebf, WcV, (void*)vtb, Bn * Sn, Dn, Dn);
  // 3) flash attention
  dim3 gA(Sn / 128, Hn, Bn);
  hipLaunchKernelGGL(attn_kernel, gA, blk, 0, stream, qb, kb, vtb, msk, Hb);
  // 4) output projection: out = Hb @ W_O^T (f32 out)
  hipLaunchKernelGGL((gemm_bf16<OUT_F32>), gP, blk, 0, stream, Hb, WObf, (void*)out, Bn * Sn, Dn, Dn);
}
